// DCNADSNet_lite_Model_18846316495128
// MI455X (gfx1250) — matrix operand order repaired and checked by reading, not compiled, not run on hardware
//
#include <hip/hip_runtime.h>
#include <hip/hip_bf16.h>
#include <math.h>

// ---------------------------------------------------------------------------
// DCN-ADS-Net-lite forward for MI455X (gfx1250, wave32, WMMA).
// ConvLSTM gate conv implemented as implicit GEMM with v_wmma_f32_16x16x32_f16:
//   M = 32 gate channels (2 tiles of 16), K = 2*8*25 = 400 (padded to 416),
//   N = 39*39 = 1521 pixels (96 tiles of 16).
// A operand: f16 weights pre-packed m-major -> 2x global_load_b128 per M tile.
// B operand: f16 zero-padded x/h images in LDS + branch-free offset table
//            (4x ds_load_b128 table fetch + 16x ds_load_u16 data).
// ---------------------------------------------------------------------------

typedef _Float16 v16h __attribute__((ext_vector_type(16)));
typedef float    v8f  __attribute__((ext_vector_type(8)));

struct U8x32 { uint4 lo; uint4 hi; };   // 32B carrier for v16h bit_cast

#define MN    39
#define NPIX  1521          // 39*39
#define PADW  43            // 39 + 2*2 halo
#define PADN  (PADW*PADW)   // 1849
#define NPLANE 17           // 16 data planes (x:8, h:8) + 1 zero plane
#define N159  159
#define NP159 25281         // 159*159
#define NP79  6241          // 79*79
#define NP78  6084          // 78*78
#define KTOT  400
#define KPAD  416           // 13 * 32

__device__ __forceinline__ float sigm(float x) { return 1.0f / (1.0f + __expf(-x)); }

// ======================= cnn1: conv1(1->4)+relu+pool (159->79) ==============
__global__ __launch_bounds__(256)
void cnn1_conv1_pool(const float* __restrict__ obs, const float* __restrict__ w1,
                     const float* __restrict__ b1, float* __restrict__ p1) {
  int b = blockIdx.x, t = blockIdx.y;
  const float* img = obs + (size_t)(b * 8 + t) * NP159;     // (B,T,159,159,1)
  float* out = p1 + (size_t)(t * 8 + b) * 4 * NP79;         // [t][b][4][79][79]
  for (int idx = threadIdx.x; idx < 4 * NP79; idx += 256) {
    int c = idx / NP79, q = idx % NP79;
    int qy = q / 79, qx = q % 79;
    float mx = -1e30f;
    for (int d = 0; d < 4; ++d) {
      int Y = 2 * qy + (d >> 1), X = 2 * qx + (d & 1);
      float acc = b1[c];
      for (int ky = 0; ky < 5; ++ky) {
        int y = Y + ky - 2; if (y < 0 || y >= N159) continue;
        for (int kx = 0; kx < 5; ++kx) {
          int x = X + kx - 2; if (x < 0 || x >= N159) continue;
          acc += w1[(c * 5 + ky) * 5 + kx] * img[y * N159 + x];
        }
      }
      mx = fmaxf(mx, fmaxf(acc, 0.0f));
    }
    out[idx] = mx;
  }
}

// ======================= cnn1: conv2(4->4)+relu+pool (79->39) ===============
__global__ __launch_bounds__(256)
void cnn1_conv2_pool(const float* __restrict__ p1, const float* __restrict__ w2,
                     const float* __restrict__ b2, float* __restrict__ p2) {
  int b = blockIdx.x, t = blockIdx.y;
  const float* ib = p1 + (size_t)(t * 8 + b) * 4 * NP79;
  float* ob = p2 + (size_t)(t * 8 + b) * 4 * NPIX;
  for (int idx = threadIdx.x; idx < 4 * NPIX; idx += 256) {
    int c = idx / NPIX, q = idx % NPIX;
    int qy = q / MN, qx = q % MN;
    float mx = -1e30f;
    for (int d = 0; d < 4; ++d) {
      int Y = 2 * qy + (d >> 1), X = 2 * qx + (d & 1);
      float acc = b2[c];
      for (int ci = 0; ci < 4; ++ci)
        for (int ky = 0; ky < 5; ++ky) {
          int y = Y + ky - 2; if (y < 0 || y >= 79) continue;
          for (int kx = 0; kx < 5; ++kx) {
            int x = X + kx - 2; if (x < 0 || x >= 79) continue;
            acc += w2[((c * 4 + ci) * 5 + ky) * 5 + kx] * ib[ci * NP79 + y * 79 + x];
          }
        }
      mx = fmaxf(mx, fmaxf(acc, 0.0f));
    }
    ob[idx] = mx;
  }
}

// ======================= cnn1: conv3(4->8) + LayerNorm ======================
__global__ __launch_bounds__(256)
void cnn1_conv3_ln(const float* __restrict__ p2, const float* __restrict__ w3,
                   const float* __restrict__ b3, const float* __restrict__ g,
                   const float* __restrict__ beta, float* __restrict__ xenc) {
  __shared__ float ybuf[8 * NPIX];
  __shared__ float red[2][256];
  int b = blockIdx.x, t = blockIdx.y, tid = threadIdx.x;
  const float* ib = p2 + (size_t)(t * 8 + b) * 4 * NPIX;
  float s = 0.0f, s2 = 0.0f;
  for (int idx = tid; idx < 8 * NPIX; idx += 256) {
    int o = idx / NPIX, p = idx % NPIX;
    int py = p / MN, px = p % MN;
    float acc = b3[o];
    for (int c = 0; c < 4; ++c)
      for (int ky = 0; ky < 5; ++ky) {
        int y = py + ky - 2; if (y < 0 || y >= MN) continue;
        for (int kx = 0; kx < 5; ++kx) {
          int x = px + kx - 2; if (x < 0 || x >= MN) continue;
          acc += w3[((o * 4 + c) * 5 + ky) * 5 + kx] * ib[c * NPIX + y * MN + x];
        }
      }
    ybuf[idx] = acc; s += acc; s2 += acc * acc;
  }
  red[0][tid] = s; red[1][tid] = s2;
  __syncthreads();
  for (int off = 128; off > 0; off >>= 1) {
    if (tid < off) { red[0][tid] += red[0][tid + off]; red[1][tid] += red[1][tid + off]; }
    __syncthreads();
  }
  float mu = red[0][0] / (8.0f * NPIX);
  float var = red[1][0] / (8.0f * NPIX) - mu * mu;
  float rs = rsqrtf(var + 1e-5f);
  float* ob = xenc + (size_t)(t * 8 + b) * 8 * NPIX;
  for (int idx = tid; idx < 8 * NPIX; idx += 256)
    ob[idx] = (ybuf[idx] - mu) * rs * g[idx] + beta[idx];
}

// ======================= lite encoder: 5 branches conv+pool+pool ============
__global__ __launch_bounds__(256)
void lite_branches(const float* __restrict__ wrf,
                   const float* __restrict__ wqi, const float* __restrict__ bqi,
                   const float* __restrict__ wqs, const float* __restrict__ bqs,
                   const float* __restrict__ wqg, const float* __restrict__ bqg,
                   const float* __restrict__ ww,  const float* __restrict__ bw,
                   const float* __restrict__ wr,  const float* __restrict__ br,
                   float* __restrict__ lpc) {
  int b = blockIdx.x, t = blockIdx.y;
  const float* base = wrf + ((size_t)b * 12 + t) * (size_t)NP159 * 29;  // (y,x,c)
  float* out = lpc + (size_t)(t * 8 + b) * 5 * NPIX;
  for (int idx = threadIdx.x; idx < 5 * NPIX; idx += 256) {
    int brn = idx / NPIX, p = idx % NPIX;
    int py = p / MN, px = p % MN;
    const float* wb; float bb; int cb, ci;
    switch (brn) {
      case 0:  wb = wqi; bb = bqi[0]; cb = 3;  ci = 4; break;
      case 1:  wb = wqs; bb = bqs[0]; cb = 12; ci = 4; break;
      case 2:  wb = wqg; bb = bqg[0]; cb = 21; ci = 4; break;
      case 3:  wb = ww;  bb = bw[0];  cb = 27; ci = 1; break;
      default: wb = wr;  bb = br[0];  cb = 28; ci = 1; break;
    }
    float mx = -1e30f;
    for (int dy = 0; dy < 4; ++dy)
      for (int dx = 0; dx < 4; ++dx) {
        int Y = 4 * py + dy, X = 4 * px + dx;   // <= 155 < 159
        float acc = bb;
        for (int c = 0; c < ci; ++c)
          for (int ky = 0; ky < 5; ++ky) {
            int y = Y + ky - 2; if (y < 0 || y >= N159) continue;
            for (int kx = 0; kx < 5; ++kx) {
              int x = X + kx - 2; if (x < 0 || x >= N159) continue;
              acc += wb[(c * 5 + ky) * 5 + kx] * base[(size_t)(y * N159 + x) * 29 + cb + c];
            }
          }
        mx = fmaxf(mx, fmaxf(acc, 0.0f));
      }
    out[idx] = mx;
  }
}

// ======================= lite encoder: LN(5ch) + conv we1(5->8)+relu ========
__global__ __launch_bounds__(256)
void lite_ln_conv1(const float* __restrict__ lpc, const float* __restrict__ g,
                   const float* __restrict__ beta, const float* __restrict__ we1,
                   const float* __restrict__ be1, float* __restrict__ enc1) {
  __shared__ float nbuf[5 * NPIX];
  __shared__ float red[2][256];
  int blk = blockIdx.x, tid = threadIdx.x;      // blk = t*8 + b
  const float* ib = lpc + (size_t)blk * 5 * NPIX;
  float s = 0.0f, s2 = 0.0f;
  for (int idx = tid; idx < 5 * NPIX; idx += 256) {
    float v = ib[idx]; nbuf[idx] = v; s += v; s2 += v * v;
  }
  red[0][tid] = s; red[1][tid] = s2;
  __syncthreads();
  for (int off = 128; off > 0; off >>= 1) {
    if (tid < off) { red[0][tid] += red[0][tid + off]; red[1][tid] += red[1][tid + off]; }
    __syncthreads();
  }
  float mu = red[0][0] / (5.0f * NPIX);
  float var = red[1][0] / (5.0f * NPIX) - mu * mu;
  float rs = rsqrtf(var + 1e-5f);
  for (int idx = tid; idx < 5 * NPIX; idx += 256)
    nbuf[idx] = (nbuf[idx] - mu) * rs * g[idx] + beta[idx];
  __syncthreads();
  float* ob = enc1 + (size_t)blk * 8 * NPIX;
  for (int idx = tid; idx < 8 * NPIX; idx += 256) {
    int o = idx / NPIX, p = idx % NPIX;
    int py = p / MN, px = p % MN;
    float acc = be1[o];
    for (int c = 0; c < 5; ++c)
      for (int ky = 0; ky < 5; ++ky) {
        int y = py + ky - 2; if (y < 0 || y >= MN) continue;
        for (int kx = 0; kx < 5; ++kx) {
          int x = px + kx - 2; if (x < 0 || x >= MN) continue;
          acc += we1[((o * 5 + c) * 5 + ky) * 5 + kx] * nbuf[c * NPIX + y * MN + x];
        }
      }
    ob[idx] = fmaxf(acc, 0.0f);
  }
}

// ======================= generic 5x5 conv on 39x39 ==========================
__global__ __launch_bounds__(256)
void conv5_39(const float* __restrict__ in, const float* __restrict__ w,
              const float* __restrict__ bias, float* __restrict__ out,
              int Cin, int Cout, int relu) {
  int blk = blockIdx.y;
  const float* ib = in + (size_t)blk * Cin * NPIX;
  float* ob = out + (size_t)blk * Cout * NPIX;
  for (int idx = blockIdx.x * blockDim.x + threadIdx.x; idx < Cout * NPIX;
       idx += gridDim.x * blockDim.x) {
    int o = idx / NPIX, p = idx % NPIX;
    int py = p / MN, px = p % MN;
    float acc = bias[o];
    for (int c = 0; c < Cin; ++c)
      for (int ky = 0; ky < 5; ++ky) {
        int y = py + ky - 2; if (y < 0 || y >= MN) continue;
        for (int kx = 0; kx < 5; ++kx) {
          int x = px + kx - 2; if (x < 0 || x >= MN) continue;
          acc += w[((o * Cin + c) * 5 + ky) * 5 + kx] * ib[c * NPIX + y * MN + x];
        }
      }
    ob[idx] = relu ? fmaxf(acc, 0.0f) : acc;
  }
}

// ========== weight pack: [wx||wh] -> f16 m-major wpkT[32][KPAD] =============
__global__ void pack_weights(const float* __restrict__ wx, const float* __restrict__ wh,
                             _Float16* __restrict__ wpkT) {
  int i = blockIdx.x * 256 + threadIdx.x;
  if (i >= 32 * KPAD) return;
  int m = i / KPAD, k = i % KPAD;
  float v = 0.0f;
  if (k < 200) v = wx[m * 200 + k];
  else if (k < KTOT) v = wh[m * 200 + (k - 200)];
  wpkT[m * KPAD + k] = (_Float16)v;
}

__global__ void zero_f(float* __restrict__ p, int n) {
  int i = blockIdx.x * 256 + threadIdx.x;
  if (i < n) p[i] = 0.0f;
}

// ======================= fused ConvLSTM step (WMMA f16) =====================
__global__ __launch_bounds__(256)
void lstm_step_wmma(const float* __restrict__ x,       // [B][8][NPIX]
                    const float* __restrict__ hin,     // [B][8][NPIX]
                    const float* __restrict__ cin,
                    const _Float16* __restrict__ wpkT, // [32][KPAD] f16, m-major
                    const float* __restrict__ bg,      // [32]
                    const float* __restrict__ wci, const float* __restrict__ wcf,
                    const float* __restrict__ wco,
                    float* __restrict__ hout, float* __restrict__ cout,
                    const float* __restrict__ xnext) {
  __shared__ int ofs[KPAD] __attribute__((aligned(16)));  // k -> base offset in pad
  __shared__ _Float16 pad[NPLANE * PADN];                 // f16 padded x,h + zero plane
  const int b = blockIdx.x;
  const int tid = threadIdx.x;
  const int lane = tid & 31;
  const int wv = tid >> 5;

  // branch-free k -> offset table; k >= KTOT points into the all-zero plane 16
  for (int k = tid; k < KPAD; k += 256) {
    int o;
    if (k < KTOT) {
      int src = k / 200, r = k % 200;
      int c = r / 25, t5 = r % 25;
      int ky = t5 / 5, kx = t5 % 5;
      o = ((src * 8 + c) * PADW + ky) * PADW + kx;
    } else {
      o = 16 * PADN;
    }
    ofs[k] = o;
  }
  // stage zero-padded f16 images of x and h; plane 16 = zeros
  for (int i = tid; i < NPLANE * PADN; i += 256) {
    int sc = i / PADN, rem = i % PADN;
    int yy = rem / PADW, xx = rem % PADW;
    int y = yy - 2, xq = xx - 2;
    float v = 0.0f;
    if (sc < 16 && y >= 0 && y < MN && xq >= 0 && xq < MN) {
      const float* sp = (sc < 8) ? (x + (size_t)b * 8 * NPIX) : (hin + (size_t)b * 8 * NPIX);
      v = sp[(sc & 7) * NPIX + y * MN + xq];
    }
    pad[i] = (_Float16)v;
  }
  __syncthreads();

  if (xnext) __builtin_prefetch(xnext + (size_t)b * 8 * NPIX + tid * 16, 0, 3);

  const int hi = (lane >= 16) ? 1 : 0;
  const int hioff = hi ? 8 : 0;
  const _Float16* a0p = wpkT + (size_t)(lane & 15) * KPAD;  // M tile 0 (gates 0..15)
  const _Float16* a1p = a0p + (size_t)16 * KPAD;            // M tile 1 (gates 16..31)

  for (int tile = wv; tile < 96; tile += 8) {
    int p = tile * 16 + (lane & 15);
    int pp = (p < NPIX) ? p : (NPIX - 1);
    int py = pp / MN, px = pp % MN;
    int pbase = py * PADW + px;

    v8f acc0 = {};  // gates 0..15 (gi, gf)
    v8f acc1 = {};  // gates 16..31 (gg, go)
    for (int k0 = 0; k0 < KPAD; k0 += 32) {
      // --- A fragments: two aligned b128 loads per M tile ---
      const uint4* pa0 = (const uint4*)(a0p + k0 + hioff);
      const uint4* pa1 = (const uint4*)(a1p + k0 + hioff);
      U8x32 ra0 = { pa0[0], pa0[2] };   // K runs [k0'..k0'+7], [k0'+16..k0'+23]
      U8x32 ra1 = { pa1[0], pa1[2] };
      v16h av0 = __builtin_bit_cast(v16h, ra0);
      v16h av1 = __builtin_bit_cast(v16h, ra1);
      // --- B fragment: 4x ds_load_b128 table + 16x ds_load_u16 data ---
      int kbase = k0 + (hi ? 8 : 0);
      const int4* pt = (const int4*)&ofs[kbase];
      int4 t0 = pt[0], t1 = pt[1], t2 = pt[4], t3 = pt[5];
      v16h bv;
      bv[0]  = pad[t0.x + pbase]; bv[1]  = pad[t0.y + pbase];
      bv[2]  = pad[t0.z + pbase]; bv[3]  = pad[t0.w + pbase];
      bv[4]  = pad[t1.x + pbase]; bv[5]  = pad[t1.y + pbase];
      bv[6]  = pad[t1.z + pbase]; bv[7]  = pad[t1.w + pbase];
      bv[8]  = pad[t2.x + pbase]; bv[9]  = pad[t2.y + pbase];
      bv[10] = pad[t2.z + pbase]; bv[11] = pad[t2.w + pbase];
      bv[12] = pad[t3.x + pbase]; bv[13] = pad[t3.y + pbase];
      bv[14] = pad[t3.z + pbase]; bv[15] = pad[t3.w + pbase];

      acc0 = __builtin_amdgcn_wmma_f32_16x16x32_f16(false, av0, false, bv,
                                                    (short)0, acc0, false, false);
      acc1 = __builtin_amdgcn_wmma_f32_16x16x32_f16(false, av1, false, bv,
                                                    (short)0, acc1, false, false);
    }

    // lanes 16..31 hold M=8..15 (gf / go) for the SAME pixel column -> pull down
    float gfv[8], gov[8];
#pragma unroll
    for (int v = 0; v < 8; ++v) {
      gfv[v] = __shfl_xor(acc0[v], 16, 32);
      gov[v] = __shfl_xor(acc1[v], 16, 32);
    }
    if (lane < 16 && p < NPIX) {
#pragma unroll
      for (int v = 0; v < 8; ++v) {
        float cprev = cin[(size_t)b * 8 * NPIX + v * NPIX + p];
        float gi = acc0[v] + bg[v];
        float gf = gfv[v] + bg[8 + v];
        float gg = acc1[v] + bg[16 + v];
        float go = gov[v] + bg[24 + v];
        float iv = sigm(gi + wci[v * NPIX + p] * cprev);
        float fv = sigm(gf + wcf[v * NPIX + p] * cprev);
        float c2 = fv * cprev + iv * tanhf(gg);
        float ov = sigm(go + wco[v * NPIX + p] * c2);
        hout[(size_t)b * 8 * NPIX + v * NPIX + p] = ov * tanhf(c2);
        cout[(size_t)b * 8 * NPIX + v * NPIX + p] = c2;
      }
    }
  }
}

// ======================= 1x1 conv + relu (encoder->decoder bridge) ==========
__global__ __launch_bounds__(256)
void conv1x1_relu(const float* __restrict__ in, const float* __restrict__ w,
                  const float* __restrict__ bias, float* __restrict__ out,
                  int Cin, int Cout) {
  int b = blockIdx.y;
  for (int idx = blockIdx.x * blockDim.x + threadIdx.x; idx < Cout * NPIX;
       idx += gridDim.x * blockDim.x) {
    int o = idx / NPIX, p = idx % NPIX;
    float acc = bias[o];
    for (int c = 0; c < Cin; ++c)
      acc += w[o * Cin + c] * in[(size_t)b * Cin * NPIX + c * NPIX + p];
    out[(size_t)b * Cout * NPIX + o * NPIX + p] = fmaxf(acc, 0.0f);
  }
}

// ======================= deformable 1x1 conv (dcn1x1) + relu ================
__global__ __launch_bounds__(256)
void dcn1x1_relu(const float* __restrict__ h, const float* __restrict__ woff,
                 const float* __restrict__ boff, const float* __restrict__ w,
                 const float* __restrict__ bias, float* __restrict__ out) {
  int b = blockIdx.x;
  const float* hb = h + (size_t)b * 8 * NPIX;
  float* ob = out + (size_t)b * 8 * NPIX;
  for (int p = blockIdx.y * blockDim.x + threadIdx.x; p < NPIX;
       p += gridDim.y * blockDim.x) {
    int py = p / MN, px = p % MN;
    float off[3];
    for (int k = 0; k < 3; ++k) {
      float a = boff[k];
      for (int c = 0; c < 8; ++c) a += woff[k * 8 + c] * hb[c * NPIX + p];
      off[k] = a;
    }
    float m = sigm(off[2]);
    float yy = (float)py + off[0];
    float xx = (float)px + off[1];
    float y0f = floorf(yy), x0f = floorf(xx);
    float wy = yy - y0f, wx = xx - x0f;
    int y0 = (int)y0f, x0 = (int)x0f;
    float smp[8];
    for (int c = 0; c < 8; ++c) {
      const float* img = hb + c * NPIX;
      float v = 0.0f;
      for (int d = 0; d < 4; ++d) {
        int iy = y0 + (d >> 1), ix = x0 + (d & 1);
        float valid = (iy >= 0 && iy < MN && ix >= 0 && ix < MN) ? 1.0f : 0.0f;
        int cy = iy < 0 ? 0 : (iy > MN - 1 ? MN - 1 : iy);
        int cx = ix < 0 ? 0 : (ix > MN - 1 ? MN - 1 : ix);
        float wgt = ((d >> 1) ? wy : 1.0f - wy) * ((d & 1) ? wx : 1.0f - wx);
        v += img[cy * MN + cx] * valid * wgt;
      }
      smp[c] = v * m;
    }
    for (int o = 0; o < 8; ++o) {
      float acc = bias[o];
      for (int c = 0; c < 8; ++c) acc += w[o * 8 + c] * smp[c];
      ob[o * NPIX + p] = fmaxf(acc, 0.0f);   // cnn3 applies relu(dcn1x1)
    }
  }
}

// ======================= deconv 39->78 (lhs_dilation=2, pad 2/3) + relu =====
__global__ __launch_bounds__(256)
void deconv78_relu(const float* __restrict__ in, const float* __restrict__ wt,
                   const float* __restrict__ bias, float* __restrict__ out) {
  int b = blockIdx.y;
  const float* ib = in + (size_t)b * 8 * NPIX;
  float* ob = out + (size_t)b * 8 * NP78;
  for (int idx = blockIdx.x * blockDim.x + threadIdx.x; idx < 8 * NP78;
       idx += gridDim.x * blockDim.x) {
    int o = idx / NP78, q = idx % NP78;
    int Y = q / 78, X = q % 78;
    float acc = bias[o];
    for (int i = 0; i < 8; ++i)
      for (int ky = 0; ky < 5; ++ky) {
        int ty = Y + ky - 2;
        if (ty < 0 || (ty & 1)) continue;
        int sy = ty >> 1; if (sy >= MN) continue;
        for (int kx = 0; kx < 5; ++kx) {
          int tx = X + kx - 2;
          if (tx < 0 || (tx & 1)) continue;
          int sx = tx >> 1; if (sx >= MN) continue;
          acc += wt[((i * 8 + o) * 5 + (4 - ky)) * 5 + (4 - kx)] * ib[i * NPIX + sy * MN + sx];
        }
      }
    ob[idx] = fmaxf(acc, 0.0f);
  }
}

// ======================= deconv 78->159 (pad 4/4) + relu + 1x1 -> output ====
__global__ __launch_bounds__(256)
void deconv159_out(const float* __restrict__ in, const float* __restrict__ wt,
                   const float* __restrict__ bias, const float* __restrict__ w2,
                   const float* __restrict__ b2, float* __restrict__ dout, int tstep) {
  int gid = blockIdx.x * blockDim.x + threadIdx.x;
  if (gid >= 8 * NP159) return;
  int b = gid / NP159, q = gid % NP159;
  int Y = q / N159, X = q % N159;
  const float* ib = in + (size_t)b * 8 * NP78;
  float pre = b2[0];
  for (int o = 0; o < 8; ++o) {
    float acc = bias[o];
    for (int i = 0; i < 8; ++i)
      for (int ky = 0; ky < 5; ++ky) {
        int ty = Y + ky - 4;
        if (ty < 0 || (ty & 1)) continue;
        int sy = ty >> 1; if (sy >= 78) continue;
        for (int kx = 0; kx < 5; ++kx) {
          int tx = X + kx - 4;
          if (tx < 0 || (tx & 1)) continue;
          int sx = tx >> 1; if (sx >= 78) continue;
          acc += wt[((i * 8 + o) * 5 + (4 - ky)) * 5 + (4 - kx)] * ib[i * NP78 + sy * 78 + sx];
        }
      }
    pre += w2[o] * fmaxf(acc, 0.0f);
  }
  // out (B,12,159,159,1)
  dout[((size_t)(b * 12 + tstep)) * NP159 + q] = pre;
}

// ===========================================================================
extern "C" void kernel_launch(void* const* d_in, const int* in_sizes, int n_in,
                              void* d_out, int out_size, void* d_ws, size_t ws_size,
                              hipStream_t stream) {
  (void)in_sizes; (void)n_in; (void)out_size; (void)ws_size;
  const float* wrf = (const float*)d_in[0];
  const float* obs = (const float*)d_in[1];
  int i = 2;
  // cnn1
  const float *c1_w1 = (const float*)d_in[i++], *c1_b1 = (const float*)d_in[i++];
  const float *c1_w2 = (const float*)d_in[i++], *c1_b2 = (const float*)d_in[i++];
  const float *c1_w3 = (const float*)d_in[i++], *c1_b3 = (const float*)d_in[i++];
  const float *c1_g  = (const float*)d_in[i++], *c1_be = (const float*)d_in[i++];
  // lstm_e
  const float *le_wx = (const float*)d_in[i++], *le_wh = (const float*)d_in[i++];
  const float *le_b  = (const float*)d_in[i++];
  const float *le_wci = (const float*)d_in[i++], *le_wcf = (const float*)d_in[i++];
  const float *le_wco = (const float*)d_in[i++];
  // enc_h / enc_c
  const float *eh_w = (const float*)d_in[i++], *eh_b = (const float*)d_in[i++];
  const float *ec_w = (const float*)d_in[i++], *ec_b = (const float*)d_in[i++];
  // lite
  const float *lt_wqi = (const float*)d_in[i++], *lt_bqi = (const float*)d_in[i++];
  const float *lt_wqs = (const float*)d_in[i++], *lt_bqs = (const float*)d_in[i++];
  const float *lt_wqg = (const float*)d_in[i++], *lt_bqg = (const float*)d_in[i++];
  const float *lt_ww  = (const float*)d_in[i++], *lt_bw  = (const float*)d_in[i++];
  const float *lt_wr  = (const float*)d_in[i++], *lt_br  = (const float*)d_in[i++];
  const float *lt_g   = (const float*)d_in[i++], *lt_be  = (const float*)d_in[i++];
  const float *lt_we1 = (const float*)d_in[i++], *lt_be1 = (const float*)d_in[i++];
  const float *lt_we2 = (const float*)d_in[i++], *lt_be2 = (const float*)d_in[i++];
  // lstm_d
  const float *ld_wx = (const float*)d_in[i++], *ld_wh = (const float*)d_in[i++];
  const float *ld_b  = (const float*)d_in[i++];
  const float *ld_wci = (const float*)d_in[i++], *ld_wcf = (const float*)d_in[i++];
  const float *ld_wco = (const float*)d_in[i++];
  // cnn3
  const float *c3_woff = (const float*)d_in[i++], *c3_boff = (const float*)d_in[i++];
  const float *c3_w  = (const float*)d_in[i++], *c3_b  = (const float*)d_in[i++];
  const float *c3_w1 = (const float*)d_in[i++], *c3_b1 = (const float*)d_in[i++];
  const float *c3_wt1 = (const float*)d_in[i++], *c3_bt1 = (const float*)d_in[i++];
  const float *c3_wt2 = (const float*)d_in[i++], *c3_bt2 = (const float*)d_in[i++];
  const float *c3_w2 = (const float*)d_in[i++], *c3_b2 = (const float*)d_in[i++];

  // ---- workspace carve-up (floats) ----
  float* ws = (float*)d_ws;
  size_t off = 0;
  auto alloc = [&](size_t n) { float* p = ws + off; off += n; return p; };
  float* P1   = alloc((size_t)8 * 8 * 4 * NP79);
  float* P2   = alloc((size_t)8 * 8 * 4 * NPIX);
  float* XENC = alloc((size_t)8 * 8 * 8 * NPIX);
  float* LPC  = alloc((size_t)12 * 8 * 5 * NPIX);
  float* ENC1 = alloc((size_t)12 * 8 * 8 * NPIX);
  float* XDEC = alloc((size_t)12 * 8 * 8 * NPIX);
  float* Hb[2] = { alloc((size_t)8 * 8 * NPIX), alloc((size_t)8 * 8 * NPIX) };
  float* Cb[2] = { alloc((size_t)8 * 8 * NPIX), alloc((size_t)8 * 8 * NPIX) };
  _Float16* WPKE = (_Float16*)alloc((size_t)32 * KPAD / 2);
  _Float16* WPKD = (_Float16*)alloc((size_t)32 * KPAD / 2);
  float* S  = alloc((size_t)8 * 8 * NPIX);
  float* A1 = alloc((size_t)8 * 8 * NPIX);
  float* D1 = alloc((size_t)8 * 8 * NP78);

  const size_t stateN = (size_t)8 * 8 * NPIX;  // per h/c buffer

  // ---- parallel pre-compute: encoder CNN, lite encoder, weight packs ----
  cnn1_conv1_pool<<<dim3(8, 8), 256, 0, stream>>>(obs, c1_w1, c1_b1, P1);
  cnn1_conv2_pool<<<dim3(8, 8), 256, 0, stream>>>(P1, c1_w2, c1_b2, P2);
  cnn1_conv3_ln<<<dim3(8, 8), 256, 0, stream>>>(P2, c1_w3, c1_b3, c1_g, c1_be, XENC);
  lite_branches<<<dim3(8, 12), 256, 0, stream>>>(wrf, lt_wqi, lt_bqi, lt_wqs, lt_bqs,
                                                 lt_wqg, lt_bqg, lt_ww, lt_bw,
                                                 lt_wr, lt_br, LPC);
  lite_ln_conv1<<<96, 256, 0, stream>>>(LPC, lt_g, lt_be, lt_we1, lt_be1, ENC1);
  conv5_39<<<dim3(8, 96), 256, 0, stream>>>(ENC1, lt_we2, lt_be2, XDEC, 8, 8, 1);
  pack_weights<<<(32 * KPAD + 255) / 256, 256, 0, stream>>>(le_wx, le_wh, WPKE);
  pack_weights<<<(32 * KPAD + 255) / 256, 256, 0, stream>>>(ld_wx, ld_wh, WPKD);
  zero_f<<<((int)stateN + 255) / 256, 256, 0, stream>>>(Hb[0], (int)stateN);
  zero_f<<<((int)stateN + 255) / 256, 256, 0, stream>>>(Cb[0], (int)stateN);

  // ---- encoder recurrence ----
  int cur = 0;
  for (int t = 0; t < 8; ++t) {
    const float* xn = (t < 7) ? (XENC + (size_t)(t + 1) * stateN) : nullptr;
    lstm_step_wmma<<<8, 256, 0, stream>>>(XENC + (size_t)t * stateN, Hb[cur], Cb[cur],
                                          WPKE, le_b, le_wci, le_wcf, le_wco,
                                          Hb[1 - cur], Cb[1 - cur], xn);
    cur ^= 1;
  }
  // bridge: relu(1x1 conv) on h and c
  conv1x1_relu<<<dim3(8, 8), 256, 0, stream>>>(Hb[cur], eh_w, eh_b, Hb[1 - cur], 8, 8);
  conv1x1_relu<<<dim3(8, 8), 256, 0, stream>>>(Cb[cur], ec_w, ec_b, Cb[1 - cur], 8, 8);
  cur ^= 1;

  // ---- decoder recurrence + cnn3 head ----
  for (int t = 0; t < 12; ++t) {
    const float* xn = (t < 11) ? (XDEC + (size_t)(t + 1) * stateN) : nullptr;
    lstm_step_wmma<<<8, 256, 0, stream>>>(XDEC + (size_t)t * stateN, Hb[cur], Cb[cur],
                                          WPKD, ld_b, ld_wci, ld_wcf, ld_wco,
                                          Hb[1 - cur], Cb[1 - cur], xn);
    cur ^= 1;
    dcn1x1_relu<<<dim3(8, 6), 256, 0, stream>>>(Hb[cur], c3_woff, c3_boff, c3_w, c3_b, S);
    conv5_39<<<dim3(8, 8), 256, 0, stream>>>(S, c3_w1, c3_b1, A1, 8, 8, 1);
    deconv78_relu<<<dim3(32, 8), 256, 0, stream>>>(A1, c3_wt1, c3_bt1, D1);
    deconv159_out<<<(8 * NP159 + 255) / 256, 256, 0, stream>>>(D1, c3_wt2, c3_bt2,
                                                               c3_w2, c3_b2,
                                                               (float*)d_out, t);
  }
}
